// ScaledDotProduct_8521215115360
// MI455X (gfx1250) — matrix operand order repaired and checked by reading, not compiled, not run on hardware
//
#include <hip/hip_runtime.h>

typedef __attribute__((ext_vector_type(16))) _Float16 v16h;
typedef __attribute__((ext_vector_type(8)))  _Float16 v8h;
typedef __attribute__((ext_vector_type(8)))  float    v8f;

#define WMMA_F16(A,B,C) __builtin_amdgcn_wmma_f32_16x16x32_f16( \
    false, (A), false, (B), (short)0, (C), false, false)

constexpr int Bc = 2, Hc = 12, Sc = 2048, Dc = 64;
constexpr int QTILE = 16;       // query rows per wave
constexpr int KTILE = 64;       // keys staged per iteration
constexpr int WAVES = 8;        // waves per workgroup
constexpr int ROWS  = QTILE * WAVES;   // 128 query rows per workgroup
constexpr int LDSP  = 80;       // padded LDS row (halves): 160B -> 32B-aligned subreads

__global__ __launch_bounds__(256, 1)
void attn_blockcausal_ghostmax(const float* __restrict__ q,
                               const float* __restrict__ k,
                               const float* __restrict__ v,
                               const int*   __restrict__ bid,
                               float*       __restrict__ out)
{
    __shared__ _Float16 ldsK[KTILE][LDSP];          // [key][d]   f16
    __shared__ _Float16 ldsV[Dc][LDSP];             // [d][key]   f16 (transposed)
    __shared__ _Float16 ldsP[WAVES][QTILE][LDSP];   // per-wave P staging
    __shared__ int      ldsKid[KTILE];
    __shared__ int      ldsKstart;

    const int b   = blockIdx.z;
    const int h   = blockIdx.y;
    const int qwg = blockIdx.x * ROWS;
    const int tid  = threadIdx.x;
    const int wave = tid >> 5;
    const int lane = tid & 31;
    const int hlf  = lane >> 4;       // 0 or 1 (lane half)
    const int l16  = lane & 15;
    const int q0   = qwg + wave * QTILE;

    // ---- lower-bound binary search: first key index in this workgroup's first block
    if (tid == 0) {
        const int* bb = bid + (size_t)b * Sc;
        const int val = bb[qwg];
        int lo = 0, hi = qwg;
        while (lo < hi) { int mid = (lo + hi) >> 1; if (bb[mid] < val) lo = mid + 1; else hi = mid; }
        ldsKstart = lo;
    }
    __syncthreads();
    const int kstart = ldsKstart & ~(KTILE - 1);
    const int kend   = qwg + ROWS;

    // ---- load Q tile in WMMA A-layout, fold in 1/sqrt(D)
    const float scale = 0.125f;
    const int   qrow  = q0 + l16;
    const float* qbase = q + (((size_t)(b * Hc + h)) * Sc + qrow) * Dc;
    v16h qa[2];
#pragma unroll
    for (int c = 0; c < 2; ++c) {
        const int o = hlf * 16;
        const float* p0 = qbase + 32 * c + o;
        const float* p1 = qbase + 32 * c + 8 + o;
#pragma unroll
        for (int i = 0; i < 8; ++i) {
            qa[c][i]     = (_Float16)(p0[i] * scale);
            qa[c][i + 8] = (_Float16)(p1[i] * scale);
        }
    }

    // per-lane block ids of this lane's 8 query rows (C-layout rows)
    int qid[8];
#pragma unroll
    for (int r = 0; r < 8; ++r) qid[r] = bid[(size_t)b * Sc + q0 + r + 8 * hlf];

    // online-softmax state: ghost logit == start with m=0, l=1
    v8f oacc[4];
#pragma unroll
    for (int i = 0; i < 4; ++i) { v8f z = {}; oacc[i] = z; }
    float m[8], l[8];
#pragma unroll
    for (int r = 0; r < 8; ++r) { m[r] = 0.0f; l[r] = 1.0f; }

    for (int kb = kstart; kb < kend; kb += KTILE) {
        // ---- stage K (key-major) and V (transposed) tiles as f16
        {
            const int key   = tid >> 2;              // 0..63
            const int dbase = (tid & 3) * 16;        // 0,16,32,48
            const float* kp = k + (((size_t)(b * Hc + h)) * Sc + kb + key) * Dc + dbase;
            const float* vp = v + (((size_t)(b * Hc + h)) * Sc + kb + key) * Dc + dbase;
#pragma unroll
            for (int i = 0; i < 16; ++i) {
                ldsK[key][dbase + i] = (_Float16)kp[i];
                ldsV[dbase + i][key] = (_Float16)vp[i];
            }
            if (tid < KTILE) ldsKid[tid] = bid[(size_t)b * Sc + kb + tid];
            if (kb + KTILE < kend) {                 // gfx1250 global_prefetch_b8
                __builtin_prefetch(kp + (size_t)KTILE * Dc, 0, 1);
                __builtin_prefetch(vp + (size_t)KTILE * Dc, 0, 1);
            }
        }
        __syncthreads();

        if (kb <= q0 + QTILE - 1) {   // wave-uniform: this tile has unmasked keys
            // ---- S = Q K^T  (16x64 in four 16x16 f32 accumulators)
            v8f s[4];
#pragma unroll
            for (int t = 0; t < 4; ++t) {
                const int key = 16 * t + l16;
                const v16h kb0 = *(const v16h*)&ldsK[key][hlf * 16];
                const v16h kb1 = *(const v16h*)&ldsK[key][32 + hlf * 16];
                v8f acc = {};
                acc = WMMA_F16(qa[0], kb0, acc);
                acc = WMMA_F16(qa[1], kb1, acc);
                s[t] = acc;
            }

            // ---- mask + per-row tile max
            float mt[8];
#pragma unroll
            for (int r = 0; r < 8; ++r) mt[r] = -1e30f;
#pragma unroll
            for (int t = 0; t < 4; ++t) {
                const int col  = kb + 16 * t + l16;
                const int kidc = ldsKid[16 * t + l16];
#pragma unroll
                for (int r = 0; r < 8; ++r) {
                    const int row = q0 + r + 8 * hlf;
                    const bool ok = (col <= row) && (kidc == qid[r]);
                    const float sv = ok ? s[t][r] : -1e30f;
                    s[t][r] = sv;
                    mt[r] = fmaxf(mt[r], sv);
                }
            }
#pragma unroll
            for (int d = 1; d < 16; d <<= 1)
#pragma unroll
                for (int r = 0; r < 8; ++r)
                    mt[r] = fmaxf(mt[r], __shfl_xor(mt[r], d, 32));

            // ---- online update (ghost already folded into m/l init)
            float alpha[8];
#pragma unroll
            for (int r = 0; r < 8; ++r) {
                const float mn = fmaxf(m[r], mt[r]);
                alpha[r] = __expf(m[r] - mn);
                m[r] = mn;
                l[r] *= alpha[r];
            }
            float ls[8];
#pragma unroll
            for (int r = 0; r < 8; ++r) ls[r] = 0.0f;
#pragma unroll
            for (int t = 0; t < 4; ++t)
#pragma unroll
                for (int r = 0; r < 8; ++r) {
                    const float p = __expf(s[t][r] - m[r]);
                    s[t][r] = p;
                    ls[r] += p;
                }
#pragma unroll
            for (int d = 1; d < 16; d <<= 1)
#pragma unroll
                for (int r = 0; r < 8; ++r)
                    ls[r] += __shfl_xor(ls[r], d, 32);
#pragma unroll
            for (int r = 0; r < 8; ++r) l[r] += ls[r];
#pragma unroll
            for (int oc = 0; oc < 4; ++oc)
#pragma unroll
                for (int r = 0; r < 8; ++r) oacc[oc][r] *= alpha[r];

            // ---- C-layout f32 -> A-layout f16 via per-wave LDS staging
#pragma unroll
            for (int t = 0; t < 4; ++t)
#pragma unroll
                for (int r = 0; r < 8; ++r)
                    ldsP[wave][r + 8 * hlf][16 * t + l16] = (_Float16)s[t][r];

            v16h pa[2];
            const _Float16* prow = &ldsP[wave][l16][0];
#pragma unroll
            for (int c = 0; c < 2; ++c) {
                const v8h lo = *(const v8h*)&prow[32 * c + 16 * hlf];
                const v8h hi = *(const v8h*)&prow[32 * c + 8 + 16 * hlf];
#pragma unroll
                for (int i = 0; i < 8; ++i) { pa[c][i] = lo[i]; pa[c][i + 8] = hi[i]; }
            }

            // ---- O += P V  (V transposed in LDS -> contiguous B-operand reads)
#pragma unroll
            for (int oc = 0; oc < 4; ++oc) {
                const int dcol = 16 * oc + l16;
                const v16h vb0 = *(const v16h*)&ldsV[dcol][hlf * 16];
                const v16h vb1 = *(const v16h*)&ldsV[dcol][32 + hlf * 16];
                oacc[oc] = WMMA_F16(pa[0], vb0, oacc[oc]);
                oacc[oc] = WMMA_F16(pa[1], vb1, oacc[oc]);
            }
        }
        __syncthreads();
    }

    // ---- epilogue: divide by ghostmax denominator, store f32
    float* obase = out + (((size_t)(b * Hc + h)) * Sc) * Dc;
#pragma unroll
    for (int oc = 0; oc < 4; ++oc)
#pragma unroll
        for (int r = 0; r < 8; ++r) {
            const int row = q0 + r + 8 * hlf;
            obase[(size_t)row * Dc + 16 * oc + l16] = oacc[oc][r] / l[r];
        }
}

extern "C" void kernel_launch(void* const* d_in, const int* in_sizes, int n_in,
                              void* d_out, int out_size, void* d_ws, size_t ws_size,
                              hipStream_t stream) {
    const float* q   = (const float*)d_in[0];
    const float* k   = (const float*)d_in[1];
    const float* v   = (const float*)d_in[2];
    const int*   bid = (const int*)d_in[3];
    float*       out = (float*)d_out;
    (void)in_sizes; (void)n_in; (void)out_size; (void)d_ws; (void)ws_size;

    dim3 grid(Sc / ROWS, Hc, Bc);   // (16, 12, 2)
    attn_blockcausal_ghostmax<<<grid, 256, 0, stream>>>(q, k, v, bid, out);
}
